// STGCNModel_35158602285061
// MI455X (gfx1250) — matrix operand order repaired and checked by reading, not compiled, not run on hardware
//
#include <hip/hip_runtime.h>
#include <hip/hip_bf16.h>
#include <math.h>

// ---------------- model dims ----------------
#define BDIM   8
#define TDIM   64
#define NNODE  256
#define FIN    16
#define HID    64
#define LSTMH  512
#define NEDGE  4096
#define BT     (BDIM*TDIM)            // 512
#define ROWS   (BT*NNODE)             // 131072
#define KBIG   (NNODE*HID)            // 16384
#define GDIM   (3*LSTMH)              // 1536

typedef __attribute__((ext_vector_type(16))) __bf16 v16bf;
typedef __attribute__((ext_vector_type(8)))  __bf16 v8bf;
typedef __attribute__((ext_vector_type(4)))  __bf16 v4bf;
typedef __attribute__((ext_vector_type(8)))  float  v8f;

static __device__ __forceinline__ __bf16 f2bf(float f) {
  unsigned int u = __builtin_bit_cast(unsigned int, f);
  unsigned int r = (u + 0x7FFFu + ((u >> 16) & 1u)) >> 16;
  return __builtin_bit_cast(__bf16, (unsigned short)r);
}

static __device__ __forceinline__ v16bf join16(v8bf lo, v8bf hi) {
  return __builtin_shufflevector(lo, hi, 0, 1, 2, 3, 4, 5, 6, 7,
                                         8, 9, 10, 11, 12, 13, 14, 15);
}

// ---------------- fp32 -> bf16 bulk conversion (4 elems/thread) ----------------
__global__ __launch_bounds__(256) void cvt_kernel(const float* __restrict__ src,
                                                  __bf16* __restrict__ dst, int n4) {
  int i = blockIdx.x * 256 + threadIdx.x;
  if (i >= n4) return;
  float4 f = ((const float4*)src)[i];
  v4bf o = { f2bf(f.x), f2bf(f.y), f2bf(f.z), f2bf(f.w) };
  ((v4bf*)dst)[i] = o;
}

// ---------------- W2 -> bf16, transposed to B-fragment-friendly layout ------------
__global__ void w2t_kernel(const float* __restrict__ W2, __bf16* __restrict__ w2tb) {
  int tid = threadIdx.x;                           // one block of 256
  for (int idx = tid; idx < HID * HID; idx += 256) {
    int i = idx / HID, j = idx % HID;              // W2[i][j] -> w2tb[j][i]
    w2tb[j * HID + i] = f2bf(W2[i * HID + j]);
  }
}

// ---------------- GCN normalization ----------------
__global__ void deg_init_kernel(float* deg) {
  deg[threadIdx.x] = 1.0f;                         // self-loop contributes 1
}

__global__ void deg_accum_kernel(const int* __restrict__ ei, float* deg) {
  int e = blockIdx.x * 256 + threadIdx.x;          // 16 blocks -> 4096
  atomicAdd(&deg[ei[NEDGE + e]], 1.0f);            // dst degree
}

__global__ void norm_kernel(const int* __restrict__ ei, const float* __restrict__ deg,
                            float* __restrict__ edgenorm, float* __restrict__ selfnorm) {
  int i = blockIdx.x * 256 + threadIdx.x;          // 4096 threads
  if (i < NEDGE) {
    int s = ei[i], d = ei[NEDGE + i];
    edgenorm[i] = rsqrtf(deg[s]) * rsqrtf(deg[d]);
  }
  if (i < NNODE) selfnorm[i] = 1.0f / deg[i];      // rsqrt(d)*rsqrt(d)
}

// ---------------- GCN layer 1 GEMM (K=16, scalar fp32) ----------------
__global__ __launch_bounds__(256) void gemm1_kernel(const float* __restrict__ x,
                                                    const float* __restrict__ W1,
                                                    float* __restrict__ xw) {
  __shared__ float Ws[FIN * HID];
  int tid = threadIdx.x;
  for (int i = tid; i < FIN * HID; i += 256) Ws[i] = W1[i];
  __syncthreads();
  size_t row = (size_t)blockIdx.x * 256 + tid;     // 512 blocks -> 131072 rows
  float xi[FIN];
#pragma unroll
  for (int f = 0; f < FIN; ++f) xi[f] = x[row * FIN + f];
  float* orow = xw + row * HID;
  for (int h = 0; h < HID; ++h) {
    float s = 0.f;
#pragma unroll
    for (int f = 0; f < FIN; ++f) s += xi[f] * Ws[f * HID + h];
    orow[h] = s;
  }
}

// ---------------- scatter-add aggregation (+bias, ReLU) -> bf16 ----------------
// grid: (BT, 2); 32 HID channels per block; LDS tile + ds_add_f32 atomics
__global__ __launch_bounds__(256) void agg_kernel(const float* __restrict__ xw,
                                                  const float* __restrict__ bias,
                                                  const int* __restrict__ ei,
                                                  const float* __restrict__ edgenorm,
                                                  const float* __restrict__ selfnorm,
                                                  __bf16* __restrict__ outp) {
  __shared__ float agg[NNODE][32];
  const int bt    = blockIdx.x;
  const int cbase = blockIdx.y * 32;
  const int tid   = threadIdx.x;                   // node id
  {
    const float sn = selfnorm[tid];
    const float* row = xw + ((size_t)bt * NNODE + tid) * HID + cbase;
#pragma unroll
    for (int c = 0; c < 32; ++c) agg[tid][c] = row[c] * sn;
  }
  __syncthreads();
  for (int e = tid; e < NEDGE; e += 256) {
    int s = ei[e], d = ei[NEDGE + e];
    float w = edgenorm[e];
    const float* row = xw + ((size_t)bt * NNODE + s) * HID + cbase;
#pragma unroll
    for (int c = 0; c < 32; ++c) atomicAdd(&agg[d][c], row[c] * w);
  }
  __syncthreads();
  {
    __bf16* orow = outp + ((size_t)bt * NNODE + tid) * HID + cbase;
#pragma unroll
    for (int c = 0; c < 32; ++c) {
      float v = agg[tid][c] + bias[cbase + c];
      orow[c] = f2bf(v > 0.f ? v : 0.f);
    }
  }
}

// ------ GCN layer 2 GEMM: [131072,64]x[64,64] via WMMA bf16 (W2 pre-transposed) ------
__global__ __launch_bounds__(256) void gemm2_kernel(const __bf16* __restrict__ h1b,
                                                    const __bf16* __restrict__ w2tb,
                                                    float* __restrict__ xw2) {
  __shared__ __bf16 Wlds[HID * HID];               // 8KB transposed W2
  const int tid = threadIdx.x;
  for (int i = tid; i < HID * HID; i += 256) Wlds[i] = w2tb[i];
  __syncthreads();
  const int lane  = tid & 31;
  const int mtile = blockIdx.x * 8 + (tid >> 5);   // 1024 blocks -> 8192 tiles
  const int m  = lane & 15;
  const int kh = lane >> 4;
  const __bf16* arow = h1b + ((size_t)mtile * 16 + m) * HID;
  v8f acc[4] = {};
#pragma unroll
  for (int kk = 0; kk < HID; kk += 32) {
    v16bf av = join16(*(const v8bf*)(arow + kk + kh * 16),
                      *(const v8bf*)(arow + kk + kh * 16 + 8));
#pragma unroll
    for (int nt = 0; nt < 4; ++nt) {
      const __bf16* wp = &Wlds[(nt * 16 + m) * HID + kk + kh * 16];
      v16bf bv = join16(*(const v8bf*)(wp), *(const v8bf*)(wp + 8));
      acc[nt] = __builtin_amdgcn_wmma_f32_16x16x32_bf16(false, av, false, bv,
                                                        (short)0, acc[nt], false, false);
    }
  }
#pragma unroll
  for (int nt = 0; nt < 4; ++nt)
#pragma unroll
    for (int i = 0; i < 8; ++i)
      xw2[((size_t)mtile * 16 + i + 8 * kh) * HID + nt * 16 + m] = acc[nt][i];
}

// ------ big GEMM: xproj[512,1536] = seq[512,16384] @ w_ih^T + b_ih ------------------
// each wave: one 16-row M tile x two adjacent 16-col N tiles (A fragment reused 2x)
__global__ __launch_bounds__(256) void gemm_big_kernel(const __bf16* __restrict__ Ab,
                                                       const __bf16* __restrict__ wihb,
                                                       const float* __restrict__ b_ih,
                                                       float* __restrict__ xproj) {
  const int tid   = threadIdx.x;
  const int lane  = tid & 31;
  const int gw    = blockIdx.x * 8 + (tid >> 5);   // 192 blocks -> 1536 waves
  const int mtile = gw / 48;                       // 32 M tiles
  const int ng    = gw % 48;                       // 48 N-tile pairs
  const int m  = lane & 15;
  const int kh = lane >> 4;
  const __bf16* arow  = Ab   + ((size_t)(mtile * 16 + m)) * KBIG;
  const __bf16* brow0 = wihb + ((size_t)(ng * 32 + m)) * KBIG;
  const __bf16* brow1 = wihb + ((size_t)(ng * 32 + 16 + m)) * KBIG;
  v8f acc0 = {}, acc1 = {};
  for (int kk = 0; kk < KBIG; kk += 32) {
    v16bf av = join16(*(const v8bf*)(arow + kk + kh * 16),
                      *(const v8bf*)(arow + kk + kh * 16 + 8));
    const __bf16* bp0 = brow0 + kk + kh * 16;
    const __bf16* bp1 = brow1 + kk + kh * 16;
    v16bf bv0 = join16(*(const v8bf*)(bp0), *(const v8bf*)(bp0 + 8));
    v16bf bv1 = join16(*(const v8bf*)(bp1), *(const v8bf*)(bp1 + 8));
    __builtin_prefetch(bp0 + 32, 0, 1);            // global_prefetch next B chunks
    __builtin_prefetch(bp1 + 32, 0, 1);
    acc0 = __builtin_amdgcn_wmma_f32_16x16x32_bf16(false, av, false, bv0,
                                                   (short)0, acc0, false, false);
    acc1 = __builtin_amdgcn_wmma_f32_16x16x32_bf16(false, av, false, bv1,
                                                   (short)0, acc1, false, false);
  }
  const int col0 = ng * 32 + m;
  const int col1 = col0 + 16;
  const float bi0 = b_ih[col0];
  const float bi1 = b_ih[col1];
#pragma unroll
  for (int i = 0; i < 8; ++i) {
    size_t r = (size_t)(mtile * 16 + i + 8 * kh);
    xproj[r * GDIM + col0] = acc0[i] + bi0;
    xproj[r * GDIM + col1] = acc1[i] + bi1;
  }
}

// ---------------- GRU recurrence + head: one persistent block ----------------
__global__ __launch_bounds__(1024) void gru_kernel(const float* __restrict__ xproj,
                                                   const __bf16* __restrict__ whhb,
                                                   const float* __restrict__ b_hh,
                                                   const float* __restrict__ Wc,
                                                   const float* __restrict__ bc,
                                                   float* __restrict__ out) {
  __shared__ float  hbuf[16][LSTMH];               // 32KB exact fp32 state
  __shared__ __bf16 hbf[16][LSTMH];                // 16KB bf16 mirror for A-fragments
  const int tid = threadIdx.x;
  for (int i = tid; i < 16 * LSTMH; i += 1024) {
    (&hbuf[0][0])[i] = 0.f;
    (&hbf[0][0])[i]  = f2bf(0.f);
  }
  __syncthreads();

  const int wave = tid >> 5;                       // 32 waves, one hidden 16-col tile each
  const int lane = tid & 31;
  const int m    = lane & 15;
  const int kh   = lane >> 4;
  const int hcol = wave * 16 + m;

  for (int t = 0; t < TDIM; ++t) {
    v8f acc0 = {}, acc1 = {}, acc2 = {};
    for (int ks = 0; ks < LSTMH / 32; ++ks) {
      const int k = ks * 32;
      v16bf av = join16(*(const v8bf*)&hbf[m][k + kh * 16],
                        *(const v8bf*)&hbf[m][k + kh * 16 + 8]);
      const __bf16* wr0 = whhb + ((size_t)(0 * LSTMH + hcol)) * LSTMH + k + kh * 16;
      const __bf16* wr1 = whhb + ((size_t)(1 * LSTMH + hcol)) * LSTMH + k + kh * 16;
      const __bf16* wr2 = whhb + ((size_t)(2 * LSTMH + hcol)) * LSTMH + k + kh * 16;
      v16bf bv0 = join16(*(const v8bf*)(wr0), *(const v8bf*)(wr0 + 8));
      v16bf bv1 = join16(*(const v8bf*)(wr1), *(const v8bf*)(wr1 + 8));
      v16bf bv2 = join16(*(const v8bf*)(wr2), *(const v8bf*)(wr2 + 8));
      acc0 = __builtin_amdgcn_wmma_f32_16x16x32_bf16(false, av, false, bv0,
                                                     (short)0, acc0, false, false);
      acc1 = __builtin_amdgcn_wmma_f32_16x16x32_bf16(false, av, false, bv1,
                                                     (short)0, acc1, false, false);
      acc2 = __builtin_amdgcn_wmma_f32_16x16x32_bf16(false, av, false, bv2,
                                                     (short)0, acc2, false, false);
    }
    __syncthreads();                               // all reads of h complete
    if (kh == 0) {                                 // rows 0..7 are the real batch
      const float bhr = b_hh[hcol];
      const float bhz = b_hh[LSTMH + hcol];
      const float bhn = b_hh[2 * LSTMH + hcol];
#pragma unroll
      for (int i = 0; i < 8; ++i) {
        const float* xp = xproj + ((size_t)i * TDIM + t) * GDIM;
        float xr = xp[hcol], xz = xp[LSTMH + hcol], xn = xp[2 * LSTMH + hcol];
        float hr = acc0[i] + bhr, hz = acc1[i] + bhz, hn = acc2[i] + bhn;
        float r  = 1.f / (1.f + __expf(-(xr + hr)));
        float z  = 1.f / (1.f + __expf(-(xz + hz)));
        float nn = tanhf(xn + r * hn);
        float hp = hbuf[i][hcol];
        float hnew = (1.f - z) * nn + z * hp;
        hbuf[i][hcol] = hnew;
        hbf[i][hcol]  = f2bf(hnew);
      }
    }
    __syncthreads();                               // writes visible before next step
  }
  if (tid < BDIM) {
    float s = bc[0];
    for (int i = 0; i < LSTMH; ++i) s += hbuf[tid][i] * Wc[i];
    out[tid] = s;
  }
}

// ---------------- launch ----------------
extern "C" void kernel_launch(void* const* d_in, const int* in_sizes, int n_in,
                              void* d_out, int out_size, void* d_ws, size_t ws_size,
                              hipStream_t stream) {
  const float* seq  = (const float*)d_in[0];
  const int*   ei   = (const int*)  d_in[1];
  const float* W1   = (const float*)d_in[2];
  const float* b1   = (const float*)d_in[3];
  const float* W2   = (const float*)d_in[4];
  const float* b2   = (const float*)d_in[5];
  const float* w_ih = (const float*)d_in[6];
  const float* w_hh = (const float*)d_in[7];
  const float* b_ih = (const float*)d_in[8];
  const float* b_hh = (const float*)d_in[9];
  const float* Wc   = (const float*)d_in[10];
  const float* bc   = (const float*)d_in[11];
  float* out = (float*)d_out;

  // ---- workspace layout (byte offsets, 256B aligned) ----
  char* base = (char*)d_ws;
  float*  deg      = (float*)(base);                         //   1 KB
  float*  selfnorm = (float*)(base + (1 << 10));             //   1 KB
  float*  edgenorm = (float*)(base + (2 << 10));             //  16 KB
  __bf16* whhb     = (__bf16*)(base + (32 << 10));           // 1.5 MB (1536*512)
  __bf16* w2tb     = (__bf16*)(base + (32 << 10) + 1572864); //   8 KB (64*64 transposed)
  float*  xproj    = (float*) (base + (32 << 10) + 1581056); //   3 MB (512*1536 f32)
  __bf16* hb       = (__bf16*)(base + (32 << 10) + 1581056 + 3145728);   // 16 MB
  char*   regionR  = base + (32 << 10) + 1581056 + 3145728
                          + (size_t)ROWS * HID * 2;          // 33.5 MB f32 xw, later
  float*  xwR      = (float*)regionR;                        //   reused as 50 MB bf16
  __bf16* wihbR    = (__bf16*)regionR;

  deg_init_kernel<<<1, 256, 0, stream>>>(deg);
  deg_accum_kernel<<<16, 256, 0, stream>>>(ei, deg);
  norm_kernel<<<16, 256, 0, stream>>>(ei, deg, edgenorm, selfnorm);

  // pre-convert recurrent + layer-2 weights to bf16 (W2 transposed)
  cvt_kernel<<<(GDIM * LSTMH / 4 + 255) / 256, 256, 0, stream>>>(w_hh, whhb, GDIM * LSTMH / 4);
  w2t_kernel<<<1, 256, 0, stream>>>(W2, w2tb);

  gemm1_kernel<<<ROWS / 256, 256, 0, stream>>>(seq, W1, xwR);
  agg_kernel<<<dim3(BT, 2), 256, 0, stream>>>(xwR, b1, ei, edgenorm, selfnorm, hb);

  gemm2_kernel<<<ROWS / 16 / 8, 256, 0, stream>>>(hb, w2tb, xwR);
  agg_kernel<<<dim3(BT, 2), 256, 0, stream>>>(xwR, b2, ei, edgenorm, selfnorm, hb);

  // xw scratch is dead now: recycle region as bf16 image of w_ih
  cvt_kernel<<<(GDIM * KBIG / 4 + 255) / 256, 256, 0, stream>>>(w_ih, wihbR, GDIM * KBIG / 4);

  gemm_big_kernel<<<(BT / 16) * (GDIM / 16) / 2 / 8, 256, 0, stream>>>(hb, wihbR, b_ih, xproj);

  gru_kernel<<<1, 1024, 0, stream>>>(xproj, whhb, b_hh, Wc, bc, out);
}
